// DeeperHNN_88295937671288
// MI455X (gfx1250) — matrix operand order repaired and checked by reading, not compiled, not run on hardware
//
#include <hip/hip_runtime.h>

// Problem constants (match reference)
#define N_NODES 100000
#define N_EDGES 20000
#define NNZ_PAIRS 800000
#define IN_DIM 384
#define HDIM 256
#define OUT_DIM 16
#define NLAYERS 4

typedef __attribute__((ext_vector_type(16))) __bf16 v16bf;
typedef __attribute__((ext_vector_type(8)))  float  v8f;

static inline int cdiv_ll(long long a, long long b) { return (int)((a + b - 1) / b); }

// ---------------------------------------------------------------- utilities

__global__ __launch_bounds__(256) void zero_kernel(float4* __restrict__ p, long long n4) {
    long long i = (long long)blockIdx.x * blockDim.x + threadIdx.x;
    if (i < n4) p[i] = make_float4(0.f, 0.f, 0.f, 0.f);
}

__global__ __launch_bounds__(256) void degree_kernel(const int* __restrict__ vidx,
                                                     const int* __restrict__ eidx,
                                                     float* __restrict__ dv,
                                                     float* __restrict__ de, int nnz) {
    int i = blockIdx.x * blockDim.x + threadIdx.x;
    if (i < nnz) {
        atomicAdd(&dv[vidx[i]], 1.0f);
        atomicAdd(&de[eidx[i]], 1.0f);
    }
}

__global__ __launch_bounds__(256) void recip_kernel(float* __restrict__ p, int n) {
    int i = blockIdx.x * blockDim.x + threadIdx.x;
    if (i < n) p[i] = 1.0f / fmaxf(p[i], 1.0f);
}

__global__ __launch_bounds__(256) void cvt_bf16_kernel(const float* __restrict__ w,
                                                       __bf16* __restrict__ o, int n) {
    int i = blockIdx.x * blockDim.x + threadIdx.x;
    if (i < n) o[i] = (__bf16)w[i];
}

// --------------------------------------------------- WMMA GEMM  C = A*B + bias
// A: [M x K] f32 row-major (converted to bf16 in-register, read exactly once)
// B: [K x NB*16] bf16 row-major (pre-converted weights, L2-resident)
// One wave32 computes a 16 x (NB*16) row-block: NB accumulators, NB WMMAs per
// K-step of 32 via v_wmma_f32_16x16x32_bf16. NB = N/16.
template <int NB>
__global__ __launch_bounds__(256) void gemm_wmma_kernel(const float* __restrict__ A,
                                                        const __bf16* __restrict__ B,
                                                        const float* __restrict__ bias,
                                                        float* __restrict__ C,
                                                        int M, int K) {
    const int N = NB * 16;
    const int lane = threadIdx.x & 31;
    const int wave = threadIdx.x >> 5;
    const long long tile = (long long)blockIdx.x * 8 + wave;   // M-tile index
    const int tilesM = M >> 4;
    if (tile >= tilesM) return;        // wave-uniform exit (EXEC stays all-1)

    const int m0   = (int)(tile << 4);
    const int nlo  = lane & 15;
    const int mrow = m0 + nlo;
    const int ks   = (lane >> 4) << 3; // +8 K-offset for lanes 16..31 (A frag)

    v8f acc[NB];
#pragma unroll
    for (int t = 0; t < NB; ++t)
#pragma unroll
        for (int r = 0; r < 8; ++r) acc[t][r] = 0.0f;

    const float* arow = A + (size_t)mrow * K;

    for (int k0 = 0; k0 < K; k0 += 32) {
        // A fragment: lane holds row mrow, K = k0 + ks + {0,2,4,6,16,18,20,22}+{0,1}
        v16bf af;
#pragma unroll
        for (int j = 0; j < 8; ++j) {
            const int kk = k0 + ks + ((j & 3) << 1) + ((j >> 2) << 4);
            const float2 a2 = *(const float2*)(arow + kk);
            af[2 * j]     = (__bf16)a2.x;
            af[2 * j + 1] = (__bf16)a2.y;
        }
        // B fragments: lane holds B column (t*16+nlo), K = k0 + ks + {0..7,16..23}
        const __bf16* brow = B + (size_t)(k0 + ks) * N + nlo;
#pragma unroll
        for (int t = 0; t < NB; ++t) {
            v16bf bf;
#pragma unroll
            for (int s = 0; s < 16; ++s)
                bf[s] = brow[(size_t)((s & 7) + ((s >> 3) << 4)) * N + t * 16];
            acc[t] = __builtin_amdgcn_wmma_f32_16x16x32_bf16(false, af, false, bf,
                                                             (short)0, acc[t], false, false);
        }
    }

    // C/D frag: VGPR r -> M = m0 + ks + r, N = t*16 + nlo
    const int mbase = m0 + ks;
#pragma unroll
    for (int t = 0; t < NB; ++t) {
        const int ncol = t * 16 + nlo;
        const float bn = bias[ncol];
#pragma unroll
        for (int r = 0; r < 8; ++r)
            C[(size_t)(mbase + r) * N + ncol] = acc[t][r] + bn;
    }
}

// ------------------------------------------------- segment-sum scatter (v2v halves)
// For pair p: dst[didx[p], :] += src[sidx[p], :] * (sscale ? sscale[sidx[p]] : 1)
__global__ __launch_bounds__(256) void scatter_kernel(const float4* __restrict__ src,
                                                      const int* __restrict__ sidx,
                                                      const int* __restrict__ didx,
                                                      const float* __restrict__ sscale,
                                                      float* __restrict__ dst,
                                                      long long total4) {
    long long i = (long long)blockIdx.x * blockDim.x + threadIdx.x;
    if (i >= total4) return;
    const int p  = (int)(i >> 6);        // HDIM/4 = 64 float4 per row
    const int f4 = (int)(i & 63);
    const int s = sidx[p];
    const int d = didx[p];
    const float4 v = src[(size_t)s * 64 + f4];
    const float sc = sscale ? sscale[s] : 1.0f;
    float* dp = dst + (size_t)d * HDIM + (f4 << 2);
    atomicAdd(dp + 0, v.x * sc);
    atomicAdd(dp + 1, v.y * sc);
    atomicAdd(dp + 2, v.z * sc);
    atomicAdd(dp + 3, v.w * sc);
}

// h = [residual ? h + : ] relu(acc * dv_inv[row])
__global__ __launch_bounds__(256) void finish_kernel(float* __restrict__ h,
                                                     const float* __restrict__ acc,
                                                     const float* __restrict__ dvinv,
                                                     int residual, long long total) {
    long long i = (long long)blockIdx.x * blockDim.x + threadIdx.x;
    if (i >= total) return;
    float v = acc[i] * dvinv[i >> 8];    // HDIM = 256
    v = fmaxf(v, 0.0f);
    h[i] = residual ? (h[i] + v) : v;
}

// out[row,:] = relu(layernorm(x[row,:]) * g + b), one block per row, H = 256
__global__ __launch_bounds__(256) void ln_relu_kernel(const float* __restrict__ x,
                                                      const float* __restrict__ g,
                                                      const float* __restrict__ b,
                                                      float* __restrict__ out) {
    const int row = blockIdx.x;
    const int t = threadIdx.x;
    const float v = x[(size_t)row * HDIM + t];

    __shared__ float sa[HDIM];
    __shared__ float sb[HDIM];
    sa[t] = v;
    sb[t] = v * v;
    __syncthreads();
#pragma unroll
    for (int s = HDIM / 2; s > 0; s >>= 1) {
        if (t < s) { sa[t] += sa[t + s]; sb[t] += sb[t + s]; }
        __syncthreads();
    }
    __shared__ float mu_s, rstd_s;
    if (t == 0) {
        const float mu  = sa[0] * (1.0f / HDIM);
        const float var = sb[0] * (1.0f / HDIM) - mu * mu;
        mu_s = mu;
        rstd_s = rsqrtf(var + 1e-5f);
    }
    __syncthreads();
    const float o = (v - mu_s) * rstd_s * g[t] + b[t];
    out[(size_t)row * HDIM + t] = fmaxf(o, 0.0f);
}

// ---------------------------------------------------------------- launcher

extern "C" void kernel_launch(void* const* d_in, const int* in_sizes, int n_in,
                              void* d_out, int out_size, void* d_ws, size_t ws_size,
                              hipStream_t stream) {
    (void)in_sizes; (void)n_in; (void)out_size; (void)ws_size;

    const float* x      = (const float*)d_in[0];
    const int*   vidx   = (const int*)  d_in[1];
    const int*   eidx   = (const int*)  d_in[2];
    const float* encW   = (const float*)d_in[3];
    const float* encB   = (const float*)d_in[4];
    const float* thetaW = (const float*)d_in[5];
    const float* thetaB = (const float*)d_in[6];
    const float* lnG    = (const float*)d_in[7];
    const float* lnB    = (const float*)d_in[8];
    const float* linW   = (const float*)d_in[9];
    const float* linB   = (const float*)d_in[10];
    float* out = (float*)d_out;

    // Workspace carve (256B aligned)
    char* w = (char*)d_ws;
    auto take = [&](size_t bytes) -> char* {
        char* p = w;
        w += (bytes + 255) & ~(size_t)255;
        return p;
    };
    float*  de      = (float*) take((size_t)N_EDGES * 4);
    float*  dv      = (float*) take((size_t)N_NODES * 4);
    float*  h       = (float*) take((size_t)N_NODES * HDIM * 4);
    float*  a       = (float*) take((size_t)N_NODES * HDIM * 4);
    float*  gbuf    = (float*) take((size_t)N_NODES * HDIM * 4);
    float*  ye      = (float*) take((size_t)N_EDGES * HDIM * 4);
    __bf16* encWb   = (__bf16*)take((size_t)IN_DIM * HDIM * 2);
    __bf16* thetaWb = (__bf16*)take((size_t)NLAYERS * HDIM * HDIM * 2);
    __bf16* linWb   = (__bf16*)take((size_t)HDIM * OUT_DIM * 2);

    const long long nodeElems = (long long)N_NODES * HDIM;
    const long long edgeElems = (long long)N_EDGES * HDIM;
    const long long scat4     = (long long)NNZ_PAIRS * (HDIM / 4);
    const int mBlocks = cdiv_ll(N_NODES / 16, 8);   // 8 waves (M-tiles) per block

    // 1) degrees -> reciprocals
    zero_kernel<<<cdiv_ll(N_EDGES / 4, 256), 256, 0, stream>>>((float4*)de, N_EDGES / 4);
    zero_kernel<<<cdiv_ll(N_NODES / 4, 256), 256, 0, stream>>>((float4*)dv, N_NODES / 4);
    degree_kernel<<<cdiv_ll(NNZ_PAIRS, 256), 256, 0, stream>>>(vidx, eidx, dv, de, NNZ_PAIRS);
    recip_kernel<<<cdiv_ll(N_EDGES, 256), 256, 0, stream>>>(de, N_EDGES);
    recip_kernel<<<cdiv_ll(N_NODES, 256), 256, 0, stream>>>(dv, N_NODES);

    // 2) weights -> bf16 once (reused by every row)
    cvt_bf16_kernel<<<cdiv_ll(IN_DIM * HDIM, 256), 256, 0, stream>>>(encW, encWb, IN_DIM * HDIM);
    cvt_bf16_kernel<<<cdiv_ll(NLAYERS * HDIM * HDIM, 256), 256, 0, stream>>>(thetaW, thetaWb,
                                                                             NLAYERS * HDIM * HDIM);
    cvt_bf16_kernel<<<cdiv_ll(HDIM * OUT_DIM, 256), 256, 0, stream>>>(linW, linWb, HDIM * OUT_DIM);

    // 3) encoder: h = x @ encW + encB   (N = 256 -> NB = 16)
    gemm_wmma_kernel<HDIM / 16><<<mBlocks, 256, 0, stream>>>(x, encWb, encB, h, N_NODES, IN_DIM);

    // 4) conv layers
    for (int i = 0; i < NLAYERS; ++i) {
        const float* Ain = h;
        if (i != 0) {
            ln_relu_kernel<<<N_NODES, HDIM, 0, stream>>>(h, lnG + i * HDIM, lnB + i * HDIM, a);
            Ain = a;
        }
        // g = Ain @ thetaW[i] + thetaB[i]
        gemm_wmma_kernel<HDIM / 16><<<mBlocks, 256, 0, stream>>>(
            Ain, thetaWb + (size_t)i * HDIM * HDIM, thetaB + i * HDIM, gbuf, N_NODES, HDIM);
        // v -> e: ye[e] += g[v]
        zero_kernel<<<cdiv_ll(edgeElems / 4, 256), 256, 0, stream>>>((float4*)ye, edgeElems / 4);
        scatter_kernel<<<cdiv_ll(scat4, 256), 256, 0, stream>>>((const float4*)gbuf, vidx, eidx,
                                                                (const float*)nullptr, ye, scat4);
        // e -> v: a[v] += ye[e] * de_inv[e]
        zero_kernel<<<cdiv_ll(nodeElems / 4, 256), 256, 0, stream>>>((float4*)a, nodeElems / 4);
        scatter_kernel<<<cdiv_ll(scat4, 256), 256, 0, stream>>>((const float4*)ye, eidx, vidx,
                                                                de, a, scat4);
        // h = (i ? h + : ) relu(a * dv_inv)
        finish_kernel<<<cdiv_ll(nodeElems, 256), 256, 0, stream>>>(h, a, dv, (i != 0) ? 1 : 0,
                                                                   nodeElems);
    }

    // 5) final: out = relu(LN(h; lnG[0], lnB[0])) @ linW + linB  (N = 16 -> NB = 1)
    ln_relu_kernel<<<N_NODES, HDIM, 0, stream>>>(h, lnG, lnB, a);
    gemm_wmma_kernel<OUT_DIM / 16><<<mBlocks, 256, 0, stream>>>(a, linWb, linB, out,
                                                                N_NODES, HDIM);
}
